// MetaRLScreener_21277267984763
// MI455X (gfx1250) — matrix operand order repaired and checked by reading, not compiled, not run on hardware
//
#include <hip/hip_runtime.h>
#include <math.h>

typedef __attribute__((ext_vector_type(16))) __bf16 v16bf;
typedef __attribute__((ext_vector_type(8)))  float  v8f;

#define NB1 1024   // blocks for GEMM/score kernel
#define NB3 1024   // blocks for softmax numerator kernel

// Branchless ELU using the native exp unit (v_exp_f32), no exec-mask branches.
__device__ __forceinline__ float eluf(float x) {
    float e = __expf(x) - 1.0f;
    return x > 0.0f ? x : e;
}

// Load one 16-element bf16 slice of a 16x32 A-tile row for this lane.
// ISA A layout (16-bit, 16x32): lanes 0-15 / 16-31 both hold M=lane&15;
// lo half-lanes: K = {0..7, 16..23}; hi half-lanes: K = {8..15, 24..31}.
__device__ __forceinline__ v16bf load_a(const float* __restrict__ row, int hi) {
    const float* p = row + (hi << 3);           // base = hi*8, 32B aligned
    float4 x0 = *(const float4*)(p);
    float4 x1 = *(const float4*)(p + 4);
    float4 y0 = *(const float4*)(p + 16);
    float4 y1 = *(const float4*)(p + 20);
    v16bf a;
    a[0]  = (__bf16)x0.x; a[1]  = (__bf16)x0.y; a[2]  = (__bf16)x0.z; a[3]  = (__bf16)x0.w;
    a[4]  = (__bf16)x1.x; a[5]  = (__bf16)x1.y; a[6]  = (__bf16)x1.z; a[7]  = (__bf16)x1.w;
    a[8]  = (__bf16)y0.x; a[9]  = (__bf16)y0.y; a[10] = (__bf16)y0.z; a[11] = (__bf16)y0.w;
    a[12] = (__bf16)y1.x; a[13] = (__bf16)y1.y; a[14] = (__bf16)y1.z; a[15] = (__bf16)y1.w;
    return a;
}

// Kernel 1: fused gather + [16x96]x[96x32] bf16 WMMA GEMM + bias + ELU +
// dot with (graph-subgraph) -> per-edge score; per-block min/max partials.
__global__ void __launch_bounds__(256)
k1_gemm_score(const float* __restrict__ node_reps,
              const float* __restrict__ edge_reps,
              const float* __restrict__ graph_rep,
              const float* __restrict__ subgraph_rep,
              const float* __restrict__ W,
              const float* __restrict__ bvec,
              const long long* __restrict__ ei0,
              const long long* __restrict__ ei1,
              float* __restrict__ sp,
              float* __restrict__ bmin,
              float* __restrict__ bmax,
              long long E)
{
    const int lane = threadIdx.x & 31;
    const int nloc = lane & 15;
    const int hi   = lane >> 4;   // which K half-slice this lane holds

    // B tiles of W (96x32) as bf16: 3 K-chunks x 2 N-halves, each 32x16.
    // ISA B layout (16-bit, 32x16): lane = column N; lo lanes K={0..7,16..23}, hi lanes K={8..15,24..31}.
    v16bf B[3][2];
    #pragma unroll
    for (int kc = 0; kc < 3; ++kc) {
        #pragma unroll
        for (int nh = 0; nh < 2; ++nh) {
            v16bf t;
            #pragma unroll
            for (int i = 0; i < 16; ++i) {
                int krow = kc * 32 + (i < 8 ? hi * 8 + i : 16 + hi * 8 + (i - 8));
                t[i] = (__bf16)W[krow * 32 + nh * 16 + nloc];
            }
            B[kc][nh] = t;
        }
    }
    const float b0 = bvec[nloc];
    const float b1 = bvec[nloc + 16];
    const float g0 = graph_rep[nloc]      - subgraph_rep[nloc];
    const float g1 = graph_rep[nloc + 16] - subgraph_rep[nloc + 16];

    float mn =  1e30f;
    float mx = -1e30f;

    const long long nTiles = (E + 15) >> 4;
    const long long wave   = ((long long)blockIdx.x * blockDim.x + threadIdx.x) >> 5;
    const long long nWaves = ((long long)gridDim.x * blockDim.x) >> 5;

    for (long long t = wave; t < nTiles; t += nWaves) {
        const long long base = t << 4;
        long long e = base + nloc;
        if (e >= E) e = E - 1;   // tail clamp: duplicates a valid row, stores guarded

        // prefetch next tile's edge row into caches
        if (t + nWaves < nTiles)
            __builtin_prefetch(edge_reps + ((t + nWaves) << 4) * 32 + nloc * 32, 0, 0);

        const long long s = ei0[e];
        const long long d = ei1[e];
        v16bf a_src = load_a(node_reps + s * 32, hi);
        v16bf a_dst = load_a(node_reps + d * 32, hi);
        v16bf a_edg = load_a(edge_reps + e * 32, hi);

        v8f c0 = {};
        v8f c1 = {};
        c0 = __builtin_amdgcn_wmma_f32_16x16x32_bf16(false, a_src, false, B[0][0], (short)0, c0, false, false);
        c0 = __builtin_amdgcn_wmma_f32_16x16x32_bf16(false, a_dst, false, B[1][0], (short)0, c0, false, false);
        c0 = __builtin_amdgcn_wmma_f32_16x16x32_bf16(false, a_edg, false, B[2][0], (short)0, c0, false, false);
        c1 = __builtin_amdgcn_wmma_f32_16x16x32_bf16(false, a_src, false, B[0][1], (short)0, c1, false, false);
        c1 = __builtin_amdgcn_wmma_f32_16x16x32_bf16(false, a_dst, false, B[1][1], (short)0, c1, false, false);
        c1 = __builtin_amdgcn_wmma_f32_16x16x32_bf16(false, a_edg, false, B[2][1], (short)0, c1, false, false);

        // Epilogue: bias + ELU + dot with g, reduce across the 16 lanes of
        // each half-wave (D layout: VGPR j holds row j (+8 for hi lanes)).
        float rows[8];
        #pragma unroll
        for (int j = 0; j < 8; ++j) {
            float p = eluf(c0[j] + b0);
            float q = eluf(c1[j] + b1);
            float v = p * g0 + q * g1;
            v += __shfl_xor(v, 1, 16);
            v += __shfl_xor(v, 2, 16);
            v += __shfl_xor(v, 4, 16);
            v += __shfl_xor(v, 8, 16);
            rows[j] = v;
            mn = fminf(mn, v);
            mx = fmaxf(mx, v);
        }
        if (nloc == 0) {
            const long long rbase = base + (long long)hi * 8;
            if (base + 16 <= E) {
                *(float4*)(sp + rbase)     = make_float4(rows[0], rows[1], rows[2], rows[3]);
                *(float4*)(sp + rbase + 4) = make_float4(rows[4], rows[5], rows[6], rows[7]);
            } else {
                for (int j = 0; j < 8; ++j)
                    if (rbase + j < E) sp[rbase + j] = rows[j];
            }
        }
    }

    // wave then block reduction of min/max
    #pragma unroll
    for (int off = 16; off >= 1; off >>= 1) {
        mn = fminf(mn, __shfl_xor(mn, off, 32));
        mx = fmaxf(mx, __shfl_xor(mx, off, 32));
    }
    __shared__ float smn[8], smx[8];
    const int w = threadIdx.x >> 5;
    if (lane == 0) { smn[w] = mn; smx[w] = mx; }
    __syncthreads();
    if (threadIdx.x == 0) {
        float m1 = smn[0], m2 = smx[0];
        for (int i = 1; i < (int)(blockDim.x >> 5); ++i) {
            m1 = fminf(m1, smn[i]);
            m2 = fmaxf(m2, smx[i]);
        }
        bmin[blockIdx.x] = m1;
        bmax[blockIdx.x] = m2;
    }
}

// Kernel 2: finalize global min/max (single block).
__global__ void __launch_bounds__(256)
k2_minmax(const float* __restrict__ bmin, const float* __restrict__ bmax,
          float* __restrict__ scal, int n)
{
    __shared__ float smn[256], smx[256];
    float mn = 1e30f, mx = -1e30f;
    for (int i = threadIdx.x; i < n; i += blockDim.x) {
        mn = fminf(mn, bmin[i]);
        mx = fmaxf(mx, bmax[i]);
    }
    smn[threadIdx.x] = mn;
    smx[threadIdx.x] = mx;
    __syncthreads();
    for (int s = 128; s > 0; s >>= 1) {
        if ((int)threadIdx.x < s) {
            smn[threadIdx.x] = fminf(smn[threadIdx.x], smn[threadIdx.x + s]);
            smx[threadIdx.x] = fmaxf(smx[threadIdx.x], smx[threadIdx.x + s]);
        }
        __syncthreads();
    }
    if (threadIdx.x == 0) { scal[0] = smn[0]; scal[1] = smx[0]; }
}

// Kernel 3: mask + exp((x - max)/T), write numerators, per-block sums.
__global__ void __launch_bounds__(256)
k3_expmask(const float* __restrict__ sp, const unsigned char* __restrict__ sel,
           const float* __restrict__ scal, float* __restrict__ out,
           float* __restrict__ bsum, long long E)
{
    const float mn  = scal[0];
    const float mx2 = scal[1] * 2.0f;
    float acc = 0.0f;
    const long long idx    = (long long)blockIdx.x * blockDim.x + threadIdx.x;
    const long long stride = (long long)gridDim.x * blockDim.x;
    const long long n4 = E >> 2;
    for (long long i = idx; i < n4; i += stride) {
        float4 x = ((const float4*)sp)[i];
        uchar4 s = ((const uchar4*)sel)[i];
        float v0 = s.x ? mn : x.x;
        float v1 = s.y ? mn : x.y;
        float v2 = s.z ? mn : x.z;
        float v3 = s.w ? mn : x.w;
        v0 = __expf(fmaf(v0, 2.0f, -mx2));   // 1/T = 2; arg <= 0 so no overflow
        v1 = __expf(fmaf(v1, 2.0f, -mx2));
        v2 = __expf(fmaf(v2, 2.0f, -mx2));
        v3 = __expf(fmaf(v3, 2.0f, -mx2));
        ((float4*)out)[i] = make_float4(v0, v1, v2, v3);
        acc += (v0 + v1) + (v2 + v3);
    }
    for (long long i = (n4 << 2) + idx; i < E; i += stride) {
        float x = sel[i] ? mn : sp[i];
        float v = __expf(fmaf(x, 2.0f, -mx2));
        out[i] = v;
        acc += v;
    }
    #pragma unroll
    for (int off = 16; off >= 1; off >>= 1)
        acc += __shfl_xor(acc, off, 32);
    __shared__ float ssum[8];
    const int w = threadIdx.x >> 5;
    if ((threadIdx.x & 31) == 0) ssum[w] = acc;
    __syncthreads();
    if (threadIdx.x == 0) {
        float tot = ssum[0];
        for (int i = 1; i < (int)(blockDim.x >> 5); ++i) tot += ssum[i];
        bsum[blockIdx.x] = tot;
    }
}

// Kernel 4: finalize softmax denominator (single block).
__global__ void __launch_bounds__(256)
k4_sum(const float* __restrict__ bsum, float* __restrict__ scal, int n)
{
    __shared__ float s[256];
    float acc = 0.0f;
    for (int i = threadIdx.x; i < n; i += blockDim.x) acc += bsum[i];
    s[threadIdx.x] = acc;
    __syncthreads();
    for (int st = 128; st > 0; st >>= 1) {
        if ((int)threadIdx.x < st) s[threadIdx.x] += s[threadIdx.x + st];
        __syncthreads();
    }
    if (threadIdx.x == 0) scal[2] = 1.0f / s[0];
}

// Kernel 5: normalize.
__global__ void __launch_bounds__(256)
k5_scale(float* __restrict__ out, const float* __restrict__ scal, long long E)
{
    const float inv = scal[2];
    const long long idx    = (long long)blockIdx.x * blockDim.x + threadIdx.x;
    const long long stride = (long long)gridDim.x * blockDim.x;
    const long long n4 = E >> 2;
    for (long long i = idx; i < n4; i += stride) {
        float4 x = ((float4*)out)[i];
        x.x *= inv; x.y *= inv; x.z *= inv; x.w *= inv;
        ((float4*)out)[i] = x;
    }
    for (long long i = (n4 << 2) + idx; i < E; i += stride)
        out[i] *= inv;
}

extern "C" void kernel_launch(void* const* d_in, const int* in_sizes, int n_in,
                              void* d_out, int out_size, void* d_ws, size_t ws_size,
                              hipStream_t stream)
{
    const float*         node_reps    = (const float*)d_in[0];
    const float*         edge_reps    = (const float*)d_in[1];
    const float*         graph_rep    = (const float*)d_in[2];
    const float*         subgraph_rep = (const float*)d_in[3];
    const float*         W            = (const float*)d_in[4];
    const float*         bvec         = (const float*)d_in[5];
    const long long*     ei           = (const long long*)d_in[6];  // int64 [2, E]
    const unsigned char* sel          = (const unsigned char*)d_in[7]; // jax bool: 1 byte
    float*               out          = (float*)d_out;

    const long long E = in_sizes[7];

    // workspace layout: [E scores][4 scalars][NB1 min][NB1 max][NB3 sums]
    float* sp   = (float*)d_ws;
    float* scal = sp + E;
    float* bmin = scal + 4;
    float* bmax = bmin + NB1;
    float* bsum = bmax + NB1;

    k1_gemm_score<<<NB1, 256, 0, stream>>>(node_reps, edge_reps, graph_rep,
                                           subgraph_rep, W, bvec,
                                           ei, ei + E, sp, bmin, bmax, E);
    k2_minmax<<<1, 256, 0, stream>>>(bmin, bmax, scal, NB1);
    k3_expmask<<<NB3, 256, 0, stream>>>(sp, sel, scal, out, bsum, E);
    k4_sum<<<1, 256, 0, stream>>>(bsum, scal, NB3);
    k5_scale<<<NB3, 256, 0, stream>>>(out, scal, E);
}
